// MoE_87308095193457
// MI455X (gfx1250) — matrix operand order repaired and checked by reading, not compiled, not run on hardware
//
#include <hip/hip_runtime.h>
#include <hip/hip_bf16.h>
#include <math.h>

// ---- problem dims (fixed by reference) ----
#define MB   4096   // batch B
#define ME   8      // experts
#define MD   1024   // in features
#define MH   4096   // hidden
#define MO   1024   // out features

// ---- tiling ----
#define TM   128    // workgroup M tile
#define TN   256    // workgroup N tile
#define TK   32     // K step
#define LDP  40     // LDS row pitch in bf16 elems (80B = 20 dwords -> conflict-free)

typedef __attribute__((ext_vector_type(16))) __bf16 v16bf;
typedef __attribute__((ext_vector_type(8)))  __bf16 v8bf;
typedef __attribute__((ext_vector_type(4)))  __bf16 v4bf;
typedef __attribute__((ext_vector_type(8)))  float  v8f;

union AFrag { v16bf v; v8bf h[2]; };
union BFrag { v16bf v; v8bf h[2]; };

// ---------------------------------------------------------------------------
// Gating: logits = x @ Wg + bg ; softmax over E=8.  One wave per batch row.
// ---------------------------------------------------------------------------
__global__ void __launch_bounds__(256)
moe_gating(const float* __restrict__ x, const float* __restrict__ Wg,
           const float* __restrict__ bg, float* __restrict__ probs)
{
    const int row  = blockIdx.x * 8 + (threadIdx.x >> 5);
    const int lane = threadIdx.x & 31;
    const float* xr = x + (size_t)row * MD;

    float acc[ME];
#pragma unroll
    for (int e = 0; e < ME; ++e) acc[e] = 0.f;

    for (int d = lane; d < MD; d += 32) {
        const float xv = xr[d];
        const float* wr = Wg + (size_t)d * ME;
#pragma unroll
        for (int e = 0; e < ME; ++e) acc[e] += xv * wr[e];
    }
#pragma unroll
    for (int off = 16; off >= 1; off >>= 1) {
#pragma unroll
        for (int e = 0; e < ME; ++e) acc[e] += __shfl_xor(acc[e], off, 32);
    }
    if (lane == 0) {
        float lg[ME], mx = -3.4e38f;
#pragma unroll
        for (int e = 0; e < ME; ++e) { lg[e] = acc[e] + bg[e]; mx = fmaxf(mx, lg[e]); }
        float s = 0.f;
#pragma unroll
        for (int e = 0; e < ME; ++e) { lg[e] = __expf(lg[e] - mx); s += lg[e]; }
        const float inv = 1.f / s;
#pragma unroll
        for (int e = 0; e < ME; ++e) probs[(size_t)row * ME + e] = lg[e] * inv;
    }
}

// ---------------------------------------------------------------------------
// GEMM1: Hout[B,H](bf16) = relu( X[B,D](f32) @ W[D,H](f32) + bias[H] )
// 128x256 tile / 8 waves in 2(M) x 4(N); wave tile 64x64 -> 16 wmma / K-step.
// ---------------------------------------------------------------------------
__global__ void __launch_bounds__(256)
moe_gemm1(const float* __restrict__ X, const float* __restrict__ W,
          const float* __restrict__ bias, __bf16* __restrict__ Hout)
{
    __shared__ __bf16 sA[TM * LDP];
    __shared__ __bf16 sB[TN * LDP];

    const int tid   = threadIdx.x;
    const int lane  = tid & 31;
    const int wid   = tid >> 5;
    const int lmod  = lane & 15;
    const int lhalf = lane >> 4;
    const int m0 = blockIdx.y * TM;
    const int n0 = blockIdx.x * TN;
    const int wm = (wid & 1) * 64;   // wave row offset in tile
    const int wn = (wid >> 1) * 64;  // wave col offset in tile

    const v8f vz = {0.f,0.f,0.f,0.f,0.f,0.f,0.f,0.f};
    v8f acc[4][4];
#pragma unroll
    for (int mi = 0; mi < 4; ++mi)
#pragma unroll
        for (int ni = 0; ni < 4; ++ni) acc[mi][ni] = vz;

    const int ac4 = (tid & 7) * 4;   // A: col quad within K-tile
    const int ar  = tid >> 3;        // A: row 0..31
    const int bn4 = (tid & 63) * 4;  // B: col quad within N-tile (0..252)
    const int bkr = tid >> 6;        // B: k row 0..3

    for (int k0 = 0; k0 < MD; k0 += TK) {
        // ---- fill A tile (fp32 -> bf16) ----
#pragma unroll
        for (int rb = 0; rb < TM; rb += 32) {
            const float4 v = *(const float4*)(X + (size_t)(m0 + rb + ar) * MD + k0 + ac4);
            v4bf b; b.x = (__bf16)v.x; b.y = (__bf16)v.y; b.z = (__bf16)v.z; b.w = (__bf16)v.w;
            *(v4bf*)(&sA[(rb + ar) * LDP + ac4]) = b;
        }
        // ---- fill B tile transposed: sB[n][k]  (fp32 -> bf16) ----
#pragma unroll
        for (int kb = 0; kb < TK; kb += 4) {
            const float4 v = *(const float4*)(W + (size_t)(k0 + kb + bkr) * MH + n0 + bn4);
            sB[(bn4 + 0) * LDP + kb + bkr] = (__bf16)v.x;
            sB[(bn4 + 1) * LDP + kb + bkr] = (__bf16)v.y;
            sB[(bn4 + 2) * LDP + kb + bkr] = (__bf16)v.z;
            sB[(bn4 + 3) * LDP + kb + bkr] = (__bf16)v.w;
        }
        __syncthreads();

        if (k0 + TK < MD) {  // prefetch next K-tile -> global_prefetch_b8
            __builtin_prefetch(X + (size_t)(m0 + ar) * MD + k0 + TK + ac4, 0, 1);
            __builtin_prefetch(W + (size_t)(k0 + TK + bkr) * MH + n0 + bn4, 0, 1);
        }

        // ---- WMMA: a-frag lanes 0-15 hold K0-7/16-23, lanes 16-31 hold K8-15/24-31 ----
        AFrag a[4];
#pragma unroll
        for (int mi = 0; mi < 4; ++mi) {
            const __bf16* p = &sA[(wm + mi * 16 + lmod) * LDP + lhalf * 8];
            a[mi].h[0] = *(const v8bf*)p;
            a[mi].h[1] = *(const v8bf*)(p + 16);
        }
#pragma unroll
        for (int ni = 0; ni < 4; ++ni) {
            const __bf16* p = &sB[(wn + ni * 16 + lmod) * LDP + lhalf * 8];
            BFrag b; b.h[0] = *(const v8bf*)p; b.h[1] = *(const v8bf*)(p + 16);
#pragma unroll
            for (int mi = 0; mi < 4; ++mi)
                acc[mi][ni] = __builtin_amdgcn_wmma_f32_16x16x32_bf16(
                    false, a[mi].v, false, b.v, (short)0, acc[mi][ni], false, false);
        }
        __syncthreads();
    }

    // ---- epilogue: +bias, relu, bf16 store.  C layout: M = r + 8*lhalf, N = lmod ----
#pragma unroll
    for (int ni = 0; ni < 4; ++ni) {
        const int n = n0 + wn + ni * 16 + lmod;
        const float bv = bias[n];
#pragma unroll
        for (int mi = 0; mi < 4; ++mi) {
#pragma unroll
            for (int r = 0; r < 8; ++r) {
                const int m = m0 + wm + mi * 16 + r + 8 * lhalf;
                float v = acc[mi][ni][r] + bv;
                v = v > 0.f ? v : 0.f;
                Hout[(size_t)m * MH + n] = (__bf16)v;
            }
        }
    }
}

// ---------------------------------------------------------------------------
// GEMM2: Out[B,O] (+)= probs[:,e] * ( Hin[B,H](bf16) @ W[H,O](f32) + bias[O] )
// A tile is pure bf16 -> use CDNA5 async global->LDS copies (ASYNCcnt path).
// ---------------------------------------------------------------------------
__global__ void __launch_bounds__(256)
moe_gemm2(const __bf16* __restrict__ Hin, const float* __restrict__ W,
          const float* __restrict__ bias, const float* __restrict__ probs,
          float* __restrict__ Out, int expert, int first)
{
    __shared__ __bf16 sA[TM * LDP];
    __shared__ __bf16 sB[TN * LDP];

    const int tid   = threadIdx.x;
    const int lane  = tid & 31;
    const int wid   = tid >> 5;
    const int lmod  = lane & 15;
    const int lhalf = lane >> 4;
    const int m0 = blockIdx.y * TM;
    const int n0 = blockIdx.x * TN;
    const int wm = (wid & 1) * 64;
    const int wn = (wid >> 1) * 64;

    const v8f vz = {0.f,0.f,0.f,0.f,0.f,0.f,0.f,0.f};
    v8f acc[4][4];
#pragma unroll
    for (int mi = 0; mi < 4; ++mi)
#pragma unroll
        for (int ni = 0; ni < 4; ++ni) acc[mi][ni] = vz;

    const int ach = (tid & 3) * 8;   // A: bf16 chunk (8 elems = 16B)
    const int ar  = tid >> 2;        // A: row 0..63
    const int bn4 = (tid & 63) * 4;
    const int bkr = tid >> 6;

    for (int k0 = 0; k0 < MH; k0 += TK) {
        // ---- A tile: bf16 copy via async global->LDS b128 (no VGPR round-trip) ----
#pragma unroll
        for (int rb = 0; rb < TM; rb += 64) {
            const __bf16* gp = Hin + (size_t)(m0 + rb + ar) * MH + k0 + ach;
            const unsigned loff = (unsigned)(unsigned long long)(&sA[(rb + ar) * LDP + ach]);
            asm volatile("global_load_async_to_lds_b128 %0, %1, off"
                         :: "v"(loff), "v"(gp) : "memory");
        }
        // ---- B tile transposed (fp32 -> bf16) ----
#pragma unroll
        for (int kb = 0; kb < TK; kb += 4) {
            const float4 v = *(const float4*)(W + (size_t)(k0 + kb + bkr) * MO + n0 + bn4);
            sB[(bn4 + 0) * LDP + kb + bkr] = (__bf16)v.x;
            sB[(bn4 + 1) * LDP + kb + bkr] = (__bf16)v.y;
            sB[(bn4 + 2) * LDP + kb + bkr] = (__bf16)v.z;
            sB[(bn4 + 3) * LDP + kb + bkr] = (__bf16)v.w;
        }
        asm volatile("s_wait_asynccnt 0x0" ::: "memory");
        __syncthreads();

        if (k0 + TK < MH) {
            __builtin_prefetch(Hin + (size_t)(m0 + ar) * MH + k0 + TK + ach, 0, 1);
            __builtin_prefetch(W + (size_t)(k0 + TK + bkr) * MO + n0 + bn4, 0, 1);
        }

        AFrag a[4];
#pragma unroll
        for (int mi = 0; mi < 4; ++mi) {
            const __bf16* p = &sA[(wm + mi * 16 + lmod) * LDP + lhalf * 8];
            a[mi].h[0] = *(const v8bf*)p;
            a[mi].h[1] = *(const v8bf*)(p + 16);
        }
#pragma unroll
        for (int ni = 0; ni < 4; ++ni) {
            const __bf16* p = &sB[(wn + ni * 16 + lmod) * LDP + lhalf * 8];
            BFrag b; b.h[0] = *(const v8bf*)p; b.h[1] = *(const v8bf*)(p + 16);
#pragma unroll
            for (int mi = 0; mi < 4; ++mi)
                acc[mi][ni] = __builtin_amdgcn_wmma_f32_16x16x32_bf16(
                    false, a[mi].v, false, b.v, (short)0, acc[mi][ni], false, false);
        }
        __syncthreads();
    }

    // ---- epilogue: weighted accumulate into Out ----
#pragma unroll
    for (int ni = 0; ni < 4; ++ni) {
        const int n = n0 + wn + ni * 16 + lmod;
        const float bv = bias[n];
#pragma unroll
        for (int mi = 0; mi < 4; ++mi) {
#pragma unroll
            for (int r = 0; r < 8; ++r) {
                const int m = m0 + wm + mi * 16 + r + 8 * lhalf;
                const float p = probs[(size_t)m * ME + expert];
                const float v = p * (acc[mi][ni][r] + bv);
                const size_t oi = (size_t)m * MO + n;
                Out[oi] = first ? v : (Out[oi] + v);
            }
        }
    }
}

// ---------------------------------------------------------------------------
extern "C" void kernel_launch(void* const* d_in, const int* in_sizes, int n_in,
                              void* d_out, int out_size, void* d_ws, size_t ws_size,
                              hipStream_t stream)
{
    const float* x  = (const float*)d_in[0];
    const float* W1 = (const float*)d_in[1];
    const float* b1 = (const float*)d_in[2];
    const float* W2 = (const float*)d_in[3];
    const float* b2 = (const float*)d_in[4];
    const float* Wg = (const float*)d_in[5];
    const float* bg = (const float*)d_in[6];
    float* out = (float*)d_out;

    float*  probs = (float*)d_ws;                                   // [B, E] f32 (128 KB)
    __bf16* hbuf  = (__bf16*)((char*)d_ws + (size_t)MB * ME * 4);   // [B, H] bf16 (32 MB)

    moe_gating<<<MB / 8, 256, 0, stream>>>(x, Wg, bg, probs);

    for (int e = 0; e < ME; ++e) {
        moe_gemm1<<<dim3(MH / TN, MB / TM), 256, 0, stream>>>(
            x, W1 + (size_t)e * MD * MH, b1 + (size_t)e * MH, hbuf);
        moe_gemm2<<<dim3(MO / TN, MB / TM), 256, 0, stream>>>(
            hbuf, W2 + (size_t)e * MH * MO, b2 + (size_t)e * MO, probs, out, e, e == 0 ? 1 : 0);
    }
}
